// SurfknnPntattn_89945205112937
// MI455X (gfx1250) — matrix operand order repaired and checked by reading, not compiled, not run on hardware
//
#include <hip/hip_runtime.h>
#include <hip/hip_bf16.h>

#define NPTS 8192
#define NB 2
#define MC 2048
#define DM 256
#define CIN 128
#define KNN 32

typedef __attribute__((ext_vector_type(16))) _Float16 v16h;
typedef __attribute__((ext_vector_type(8)))  _Float16 v8h;
typedef __attribute__((ext_vector_type(8)))  float    v8f;

// ---------------- prep: transpose xyz -> pT (x,y,z,|p|^2), fea -> fT/fTh ----------------
__global__ void k_prep_pt(const float* __restrict__ xyz, float4* __restrict__ pT) {
  int idx = blockIdx.x * 256 + threadIdx.x;
  if (idx >= NB * NPTS) return;
  int b = idx >> 13, n = idx & (NPTS - 1);
  const float* base = xyz + (size_t)b * 3 * NPTS;
  float x = base[n], y = base[NPTS + n], z = base[2 * NPTS + n];
  pT[idx] = make_float4(x, y, z, x * x + y * y + z * z);
}

__global__ void k_prep_f(const float* __restrict__ fea, float* __restrict__ fT,
                         _Float16* __restrict__ fTh) {
  size_t idx = (size_t)blockIdx.x * 256 + threadIdx.x;  // < NB*NPTS*CIN
  int c = (int)(idx & (CIN - 1));
  size_t bn = idx >> 7;
  int b = (int)(bn >> 13), n = (int)(bn & (NPTS - 1));
  float v = fea[((size_t)b * CIN + c) * NPTS + n];
  fT[idx] = v;
  fTh[idx] = (_Float16)v;
}

// WqT[n,k]=Wq[k,n] (256x128); WkP = Wk padded to 144x256; WvT[n,k]=Wv[k,n] padded (256x160)
__global__ void k_prep_w(const float* __restrict__ Wq, const float* __restrict__ Wk,
                         const float* __restrict__ Wv, _Float16* __restrict__ WqT,
                         _Float16* __restrict__ WkP, _Float16* __restrict__ WvT) {
  int idx = blockIdx.x * 256 + threadIdx.x;
  if (idx < 32768) {
    int n = idx >> 7, k = idx & 127;
    WqT[idx] = (_Float16)Wq[k * 256 + n];
    return;
  }
  int i2 = idx - 32768;
  if (i2 < 36864) {
    int r = i2 >> 8, c = i2 & 255;
    WkP[i2] = (_Float16)(r < 131 ? Wk[r * 256 + c] : 0.f);
    return;
  }
  int i3 = i2 - 36864;
  if (i3 < 40960) {
    int n = i3 / 160, k = i3 % 160;
    WvT[i3] = (_Float16)(k < 131 ? Wv[k * 256 + n] : 0.f);
  }
}

// ---------------- one-hop KNN (k=8, self included), reference distance formula ----------
__global__ void k_knn8(const float4* __restrict__ pT, int* __restrict__ A8) {
  int b = blockIdx.y;
  int n = blockIdx.x * 128 + threadIdx.x;
  float4 pn = pT[b * NPTS + n];
  float bd[8]; int bi[8];
#pragma unroll
  for (int i = 0; i < 8; ++i) { bd[i] = 3.4e38f; bi[i] = 0; }
  __shared__ float4 tile[128];
  for (int t0 = 0; t0 < NPTS; t0 += 128) {
    __syncthreads();
    tile[threadIdx.x] = pT[b * NPTS + t0 + threadIdx.x];
    __syncthreads();
    for (int j = 0; j < 128; ++j) {
      float4 pm = tile[j];
      float d = pn.w + pm.w - 2.f * (pn.x * pm.x + pn.y * pm.y + pn.z * pm.z);
      if (d < bd[7]) {
        float dd = d; int mm = t0 + j;
#pragma unroll
        for (int s = 0; s < 8; ++s) {
          if (dd < bd[s]) { float tf = bd[s]; int ti = bi[s]; bd[s] = dd; bi[s] = mm; dd = tf; mm = ti; }
        }
      }
    }
  }
  int* out = A8 + ((size_t)(b * NPTS) + n) * 8;
#pragma unroll
  for (int i = 0; i < 8; ++i) out[i] = bi[i];
}

// ------------- surface KNN: 64 cands = A[A[n]], sort, dedupe->inf, stable top-32 --------
__global__ void k_surface32(const float4* __restrict__ pT, const int* __restrict__ A8,
                            int* __restrict__ nbr) {
  __shared__ unsigned csL[64 * 64];
  __shared__ unsigned long long keyL[64 * 64];
  int t = threadIdx.x;
  int pid = blockIdx.x * 64 + t;
  int b = pid >> 13, n = pid & (NPTS - 1);
  const int* Ab = A8 + (size_t)b * NPTS * 8;
#pragma unroll
  for (int c = 0; c < 8; ++c) {
    int ac = Ab[n * 8 + c];
#pragma unroll
    for (int j = 0; j < 8; ++j) csL[(c * 8 + j) * 64 + t] = (unsigned)Ab[ac * 8 + j];
  }
  for (int i = 1; i < 64; ++i) {  // jnp.sort ascending
    unsigned v = csL[i * 64 + t];
    int j = i;
    while (j > 0 && csL[(j - 1) * 64 + t] > v) { csL[j * 64 + t] = csL[(j - 1) * 64 + t]; --j; }
    csL[j * 64 + t] = v;
  }
  float4 pn = pT[b * NPTS + n];
  unsigned prev = 0xffffffffu;
  for (int i = 0; i < 64; ++i) {
    unsigned u = csL[i * 64 + t];
    bool dup = (i > 0) && (u == prev);
    prev = u;
    float4 pm = pT[b * NPTS + (int)u];
    float dx = pm.x - pn.x, dy = pm.y - pn.y, dz = pm.z - pn.z;
    float d = dx * dx + dy * dy + dz * dz;
    unsigned fb = dup ? 0x7f800000u : __float_as_uint(d);  // d>=0 -> bits monotonic
    keyL[i * 64 + t] = ((unsigned long long)fb << 32) | ((unsigned long long)(unsigned)i << 16) |
                       (unsigned long long)u;
  }
  for (int i = 1; i < 64; ++i) {  // stable (dist, pos) sort == top_k(-d) semantics
    unsigned long long v = keyL[i * 64 + t];
    int j = i;
    while (j > 0 && keyL[(j - 1) * 64 + t] > v) { keyL[j * 64 + t] = keyL[(j - 1) * 64 + t]; --j; }
    keyL[j * 64 + t] = v;
  }
  int* out = nbr + (size_t)pid * KNN;
  for (int k = 0; k < KNN; ++k) out[k] = (int)(keyL[k * 64 + t] & 0xffffull);
}

// ---------------- FPS: one block per batch, register-resident dists ----------------
__global__ void __launch_bounds__(1024) k_fps(const float4* __restrict__ pT,
                                              int* __restrict__ fidx) {
  int b = blockIdx.x, t = threadIdx.x;
  float4 P[8]; float D[8];
#pragma unroll
  for (int j = 0; j < 8; ++j) { P[j] = pT[b * NPTS + t + j * 1024]; D[j] = 1e10f; }
  __shared__ float sv[32];
  __shared__ int si[32];
  __shared__ int sfar;
  int far = 0;
  int lane = t & 31, wid = t >> 5;
  for (int m = 0; m < MC; ++m) {
    if (t == 0) fidx[b * MC + m] = far;
    float4 c = pT[b * NPTS + far];
    float bv = -1.f; int bidx = 0;
#pragma unroll
    for (int j = 0; j < 8; ++j) {
      float dx = P[j].x - c.x, dy = P[j].y - c.y, dz = P[j].z - c.z;
      float dd = dx * dx + dy * dy + dz * dz;
      D[j] = fminf(D[j], dd);
      int gi = t + j * 1024;
      if (D[j] > bv) { bv = D[j]; bidx = gi; }  // strict > : first max wins (gi increasing)
    }
#pragma unroll
    for (int off = 16; off > 0; off >>= 1) {
      float v2 = __shfl_down(bv, off, 32);
      int i2 = __shfl_down(bidx, off, 32);
      if (v2 > bv || (v2 == bv && i2 < bidx)) { bv = v2; bidx = i2; }
    }
    if (lane == 0) { sv[wid] = bv; si[wid] = bidx; }
    __syncthreads();
    if (wid == 0) {
      bv = sv[lane]; bidx = si[lane];
#pragma unroll
      for (int off = 16; off > 0; off >>= 1) {
        float v2 = __shfl_down(bv, off, 32);
        int i2 = __shfl_down(bidx, off, 32);
        if (v2 > bv || (v2 == bv && i2 < bidx)) { bv = v2; bidx = i2; }
      }
      if (lane == 0) sfar = bidx;
    }
    __syncthreads();
    far = sfar;
  }
}

// ---------------- gather centers: cfea (f16) + cxyz output (transposed) ----------------
__global__ void k_gather(const float4* __restrict__ pT, const _Float16* __restrict__ fTh,
                         const int* __restrict__ fidx, _Float16* __restrict__ cfeaH,
                         float* __restrict__ outCxyz) {
  int center = blockIdx.x, c = threadIdx.x;  // 128 threads
  int b = center >> 11, m = center & (MC - 1);
  int n = fidx[center];
  cfeaH[(size_t)center * CIN + c] = fTh[((size_t)b * NPTS + n) * CIN + c];
  if (c < 3) {
    float4 p = pT[b * NPTS + n];
    outCxyz[(b * 3 + c) * MC + m] = (&p.x)[c];
  }
}

// ---------------- WMMA f16 GEMM: C[M,N] = A[M,K] * Bt[N,K]^T (+bias) ----------------
// MODE 0: f32 row-major out; MODE 1: f16 row-major out; MODE 2: scatter transposed to nf
template <int MODE>
__global__ void k_wmma_gemm(const _Float16* __restrict__ A, const _Float16* __restrict__ Bt,
                            void* __restrict__ C, const float* __restrict__ bias, int Mdim,
                            int Ndim, int Kdim) {
  int wid = threadIdx.x >> 5, lane = threadIdx.x & 31;
  int tile = blockIdx.x * 4 + wid;
  int ntn = Ndim >> 4;
  if (tile >= (Mdim >> 4) * ntn) return;
  int tm = tile / ntn, tn = tile % ntn;
  int hl = lane >> 4, l15 = lane & 15;
  const _Float16* pA = A + (size_t)(tm * 16 + l15) * Kdim;   // A row = M index (ISA A layout)
  const _Float16* pB = Bt + (size_t)(tn * 16 + l15) * Kdim;  // Bt row = N index (ISA B layout)
  v8f acc = {0.f, 0.f, 0.f, 0.f, 0.f, 0.f, 0.f, 0.f};
  for (int k = 0; k < Kdim; k += 32) {
    v8h alo = *(const v8h*)(pA + k + 8 * hl);        // K = 8h .. 8h+7
    v8h ahi = *(const v8h*)(pA + k + 16 + 8 * hl);   // K = 16+8h .. 16+8h+7
    v8h blo = *(const v8h*)(pB + k + 8 * hl);        // K = 8h .. 8h+7
    v8h bhi = *(const v8h*)(pB + k + 16 + 8 * hl);   // K = 16+8h .. 16+8h+7
    v16h a, bb;
#pragma unroll
    for (int i = 0; i < 8; ++i) { a[i] = alo[i]; a[i + 8] = ahi[i]; bb[i] = blo[i]; bb[i + 8] = bhi[i]; }
    acc = __builtin_amdgcn_wmma_f32_16x16x32_f16(false, a, false, bb, (short)0, acc, false, false);
  }
  int col = tn * 16 + l15;
  float bc = bias ? bias[col] : 0.f;
#pragma unroll
  for (int r = 0; r < 8; ++r) {  // C layout: VGPR r -> M = r + 8*hl, N = l15
    int row = tm * 16 + 8 * hl + r;
    float v = acc[r] + bc;
    if (MODE == 0) ((float*)C)[(size_t)row * Ndim + col] = v;
    else if (MODE == 1) ((_Float16*)C)[(size_t)row * Ndim + col] = (_Float16)v;
    else {
      int bb2 = row >> 11, mm = row & (MC - 1);
      ((float*)C)[((size_t)(bb2 * DM + col)) * MC + mm] = v;  // nf transposed [B,256,M]
    }
  }
}

// -------- scores = gfea . tq, softmax(32), u = attn^T @ gfea (f16, padded to 160) --------
__global__ void k_attn(const float4* __restrict__ pT, const float* __restrict__ fT,
                       const int* __restrict__ fidx, const int* __restrict__ nbr,
                       const float* __restrict__ tq, _Float16* __restrict__ uH) {
  __shared__ float s_tq[4][144];
  __shared__ float s_at[4][32];
  __shared__ int s_gk[4][32];
  int wid = threadIdx.x >> 5, lane = threadIdx.x & 31;
  int center = blockIdx.x * 4 + wid;
  int b = center >> 11;
  int n = fidx[center];
  for (int c = lane; c < 144; c += 32) s_tq[wid][c] = tq[(size_t)center * 144 + c];
  int gk = nbr[((size_t)b * NPTS + n) * KNN + lane];
  s_gk[wid][lane] = gk;
  __syncthreads();
  const float* frow = fT + ((size_t)b * NPTS + gk) * CIN;
  float s = 0.f;
  for (int c = 0; c < CIN; ++c) s += frow[c] * s_tq[wid][c];
  float4 pg = pT[b * NPTS + gk], pc = pT[b * NPTS + n];
  s += (pg.x - pc.x) * s_tq[wid][128] + (pg.y - pc.y) * s_tq[wid][129] +
       (pg.z - pc.z) * s_tq[wid][130];
  s *= 0.0625f;  // 1/sqrt(256)
  float mx = s;
#pragma unroll
  for (int off = 16; off > 0; off >>= 1) mx = fmaxf(mx, __shfl_xor(mx, off, 32));
  float e = expf(s - mx);
  float sum = e;
#pragma unroll
  for (int off = 16; off > 0; off >>= 1) sum += __shfl_xor(sum, off, 32);
  s_at[wid][lane] = e / sum;
  __syncthreads();
  for (int c = lane; c < 160; c += 32) {
    float acc = 0.f;
    if (c < CIN) {
      for (int k = 0; k < 32; ++k)
        acc += s_at[wid][k] * fT[((size_t)b * NPTS + s_gk[wid][k]) * CIN + c];
    } else if (c < 131) {
      int d = c - 128;
      float pcd = (&pc.x)[d];
      for (int k = 0; k < 32; ++k) {
        float4 pk = pT[b * NPTS + s_gk[wid][k]];
        acc += s_at[wid][k] * ((&pk.x)[d] - pcd);
      }
    }
    uH[(size_t)center * 160 + c] = (_Float16)acc;
  }
}

extern "C" void kernel_launch(void* const* d_in, const int* in_sizes, int n_in, void* d_out,
                              int out_size, void* d_ws, size_t ws_size, hipStream_t stream) {
  const float* xyz = (const float*)d_in[0];
  const float* fea = (const float*)d_in[1];
  const float* Wq = (const float*)d_in[2];
  const float* bq = (const float*)d_in[3];
  const float* Wk = (const float*)d_in[4];
  // d_in[5] = bk: q.bk is constant over the 32 keys -> softmax-invariant, dropped exactly.
  const float* Wv = (const float*)d_in[6];
  const float* bv = (const float*)d_in[7];

  char* w = (char*)d_ws;
  auto alloc = [&](size_t bytes) {
    char* p = w;
    w += (bytes + 255) & ~(size_t)255;
    return p;
  };
  float4* pT = (float4*)alloc(sizeof(float4) * NB * NPTS);
  float* fT = (float*)alloc(sizeof(float) * NB * NPTS * CIN);
  _Float16* fTh = (_Float16*)alloc(sizeof(_Float16) * NB * NPTS * CIN);
  int* A8 = (int*)alloc(sizeof(int) * NB * NPTS * 8);
  int* nbr = (int*)alloc(sizeof(int) * NB * NPTS * KNN);
  int* fidx = (int*)alloc(sizeof(int) * NB * MC);
  _Float16* cfeaH = (_Float16*)alloc(sizeof(_Float16) * NB * MC * CIN);
  _Float16* Qh = (_Float16*)alloc(sizeof(_Float16) * NB * MC * DM);
  float* tqB = (float*)alloc(sizeof(float) * NB * MC * 144);
  _Float16* uH = (_Float16*)alloc(sizeof(_Float16) * NB * MC * 160);
  _Float16* WqT = (_Float16*)alloc(sizeof(_Float16) * DM * CIN);
  _Float16* WkP = (_Float16*)alloc(sizeof(_Float16) * 144 * DM);
  _Float16* WvT = (_Float16*)alloc(sizeof(_Float16) * DM * 160);
  (void)ws_size; (void)in_sizes; (void)n_in; (void)out_size;

  float* outCxyz = (float*)d_out;                 // [B,3,2048]
  float* outNf = (float*)d_out + NB * 3 * MC;     // [B,256,2048]

  k_prep_pt<<<(NB * NPTS) / 256, 256, 0, stream>>>(xyz, pT);
  k_prep_f<<<(NB * NPTS * CIN) / 256, 256, 0, stream>>>(fea, fT, fTh);
  k_prep_w<<<432, 256, 0, stream>>>(Wq, Wk, Wv, WqT, WkP, WvT);
  k_knn8<<<dim3(NPTS / 128, NB), 128, 0, stream>>>(pT, A8);
  k_surface32<<<(NB * NPTS) / 64, 64, 0, stream>>>(pT, A8, nbr);
  k_fps<<<NB, 1024, 0, stream>>>(pT, fidx);
  k_gather<<<NB * MC, CIN, 0, stream>>>(pT, fTh, fidx, cfeaH, outCxyz);
  // Q = cfea @ Wq + bq  (f16 out feeds next GEMM)
  k_wmma_gemm<1><<<(NB * MC / 16) * (DM / 16) / 4, 128, 0, stream>>>(cfeaH, WqT, (void*)Qh, bq,
                                                                     NB * MC, DM, CIN);
  // tq = Q @ Wk^T   (N padded 131->144)
  k_wmma_gemm<0><<<(NB * MC / 16) * (144 / 16) / 4, 128, 0, stream>>>(Qh, WkP, (void*)tqB,
                                                                      nullptr, NB * MC, 144, DM);
  k_attn<<<NB * MC / 4, 128, 0, stream>>>(pT, fT, fidx, nbr, tqB, uH);
  // nf = u @ Wv + bv, scattered transposed into d_out   (K padded 131->160 with zeros)
  k_wmma_gemm<2><<<(NB * MC / 16) * (DM / 16) / 4, 128, 0, stream>>>(uH, WvT, (void*)outNf, bv,
                                                                     NB * MC, DM, 160);
}
